// EvoMoERouter_56710748176497
// MI455X (gfx1250) — matrix operand order repaired and checked by reading, not compiled, not run on hardware
//
#include <hip/hip_runtime.h>
#include <hip/hip_bf16.h>

#define TOKENS 8192     // B*S = 4*2048
#define DDIM   1024
#define FDIM   1024
#define NEXP   8

typedef __attribute__((ext_vector_type(16))) __bf16 v16bf;
typedef __attribute__((ext_vector_type(8)))  __bf16 v8bf;
typedef __attribute__((ext_vector_type(4)))  __bf16 v4bf;
typedef __attribute__((ext_vector_type(8)))  float  v8f;

// ------------- workspace element offsets (4-byte elements) -------------
#define WS_COUNTS   0                  // 8  ints
#define WS_CURSOR   8                  // 8  ints
#define WS_OFFS     16                 // 9  ints
#define WS_E1       32                 // TOKENS ints
#define WS_E2       (32 + TOKENS)      // TOKENS ints
#define WS_W1       (32 + 2*TOKENS)    // TOKENS floats
#define WS_W2       (32 + 3*TOKENS)    // TOKENS floats
#define WS_TOKLIST  (32 + 4*TOKENS)    // 2*TOKENS ints
#define WS_WLIST    (32 + 6*TOKENS)    // 2*TOKENS floats
#define WS_PSUM     (32 + 8*TOKENS)    // 28 floats

// ------------- output element offsets ----------------------------------
#define OUT_FINAL 0
#define OUT_RW    (TOKENS*FDIM)
#define OUT_UTIL  (OUT_RW + TOKENS*NEXP)
#define OUT_LBL   (OUT_UTIL + NEXP)
#define OUT_DIV   (OUT_LBL + 1)

// LDS tile geometry: 128 rows x 64 K, rows padded to 72 bf16 (36 dwords,
// gcd(36,64)=4 -> the 16 fragment row-starts are distinct mod 64 banks).
#define BK        64
#define LDS_STRIDE 72
#define KTILES    (DDIM / BK)          // 16

// =======================================================================
__global__ void moe_zero(int* wsi) {
    int t = threadIdx.x;
    if (t < 16) wsi[t] = 0;   // counts + cursors
}

// One wave (32 lanes) per token: logits, softmax, top-2, renorm.
__global__ void __launch_bounds__(256) moe_router(
    const float* __restrict__ x, const float* __restrict__ rw,
    const float* __restrict__ rb, float* __restrict__ out_rw,
    int* __restrict__ wsi, float* __restrict__ wsf)
{
    __shared__ float xs[8][DDIM];
    __shared__ float lg[8][8];
    int tid  = threadIdx.x;
    int lane = tid & 31, wv = tid >> 5;
    int tok  = blockIdx.x * 8 + wv;

    const float4* xr  = (const float4*)(x + (size_t)tok * DDIM);
    float4*       xs4 = (float4*)&xs[wv][0];
#pragma unroll
    for (int i = 0; i < 8; ++i) xs4[i*32 + lane] = xr[i*32 + lane];
    __syncthreads();

    int e = lane & 7, q = lane >> 3;          // 8 experts x 4 quarters
    const float4* wr4 = (const float4*)(rw + (size_t)e * DDIM);
    float acc = 0.f;
#pragma unroll 8
    for (int i = 0; i < 64; ++i) {
        float4 a = xs4[q*64 + i];
        float4 b = wr4[q*64 + i];
        acc += a.x*b.x + a.y*b.y + a.z*b.z + a.w*b.w;
    }
    acc += __shfl_xor(acc, 8, 32);
    acc += __shfl_xor(acc, 16, 32);
    if (lane < 8) lg[wv][lane] = acc + rb[lane];
    __syncthreads();

    if (lane == 0) {
        float l[8], p[8];
        float mx = -3.4e38f;
#pragma unroll
        for (int i = 0; i < 8; ++i) { l[i] = lg[wv][i]; mx = fmaxf(mx, l[i]); }
        float s = 0.f;
#pragma unroll
        for (int i = 0; i < 8; ++i) { p[i] = __expf(l[i] - mx); s += p[i]; }
        float inv = 1.f / s;
#pragma unroll
        for (int i = 0; i < 8; ++i) { p[i] *= inv; out_rw[(size_t)tok*8 + i] = p[i]; }
        // top-2, earliest index wins ties (matches lax.top_k)
        int i1 = 0; float v1 = p[0];
#pragma unroll
        for (int i = 1; i < 8; ++i) if (p[i] > v1) { v1 = p[i]; i1 = i; }
        int i2 = -1; float v2 = -1.f;
#pragma unroll
        for (int i = 0; i < 8; ++i) if (i != i1 && p[i] > v2) { v2 = p[i]; i2 = i; }
        float rn = 1.f / (v1 + v2);
        wsi[WS_E1 + tok] = i1; wsi[WS_E2 + tok] = i2;
        wsf[WS_W1 + tok] = v1 * rn; wsf[WS_W2 + tok] = v2 * rn;
        atomicAdd(&wsi[WS_COUNTS + i1], 1);
        atomicAdd(&wsi[WS_COUNTS + i2], 1);
    }
}

__global__ void moe_offsets(int* wsi) {
    if (threadIdx.x == 0) {
        int o = 0; wsi[WS_OFFS] = 0;
        for (int e = 0; e < NEXP; ++e) { o += wsi[WS_COUNTS + e]; wsi[WS_OFFS + e + 1] = o; }
    }
}

__global__ void moe_scatter(int* wsi, float* wsf) {
    int t = blockIdx.x * 256 + threadIdx.x;
    if (t >= TOKENS) return;
    int e1 = wsi[WS_E1 + t], e2 = wsi[WS_E2 + t];
    float w1 = wsf[WS_W1 + t], w2 = wsf[WS_W2 + t];
    int p1 = atomicAdd(&wsi[WS_CURSOR + e1], 1);
    int i1 = wsi[WS_OFFS + e1] + p1;
    wsi[WS_TOKLIST + i1] = t; wsf[WS_WLIST + i1] = w1;
    int p2 = atomicAdd(&wsi[WS_CURSOR + e2], 1);
    int i2 = wsi[WS_OFFS + e2] + p2;
    wsi[WS_TOKLIST + i2] = t; wsf[WS_WLIST + i2] = w2;
}

// out[t,f] = w1*b[e1,f] + w2*b[e2,f]  (bias term of the combine)
__global__ void __launch_bounds__(256) moe_bias_init(
    const float* __restrict__ eb, const int* __restrict__ wsi,
    const float* __restrict__ wsf, float* __restrict__ out)
{
    int t  = blockIdx.x;
    int e1 = wsi[WS_E1 + t], e2 = wsi[WS_E2 + t];
    float w1 = wsf[WS_W1 + t], w2 = wsf[WS_W2 + t];
    const float4* b1 = (const float4*)(eb + (size_t)e1 * FDIM);
    const float4* b2 = (const float4*)(eb + (size_t)e2 * FDIM);
    float4* o = (float4*)(out + (size_t)t * FDIM);
    int i = threadIdx.x;                     // FDIM/4 == 256
    float4 a = b1[i], b = b2[i], r;
    r.x = w1*a.x + w2*b.x; r.y = w1*a.y + w2*b.y;
    r.z = w1*a.z + w2*b.z; r.w = w1*a.w + w2*b.w;
    o[i] = r;
}

// Grouped GEMM, one expert per blockIdx.z. 128x128 tile, BK=64, bf16 WMMA,
// double-buffered LDS (one barrier per K-tile covers 16 WMMAs).
__global__ void __launch_bounds__(256) moe_gemm(
    const float* __restrict__ x, const float* __restrict__ ew,
    const int* __restrict__ wsi, const float* __restrict__ wsf,
    float* __restrict__ out)
{
    int expert = blockIdx.z;
    int base = wsi[WS_OFFS + expert];
    int cnt  = wsi[WS_OFFS + expert + 1] - base;
    int mBase = blockIdx.x * 128;
    if (mBase >= cnt) return;
    int nBase = blockIdx.y * 128;

    __shared__ __align__(16) __bf16 As[2][128 * LDS_STRIDE];
    __shared__ __align__(16) __bf16 Bs[2][128 * LDS_STRIDE];
    __shared__ int   toks[128];
    __shared__ float wts[128];

    int tid = threadIdx.x;
    if (tid < 128) {
        int gr = mBase + tid;
        bool v = gr < cnt;
        toks[tid] = v ? wsi[WS_TOKLIST + base + gr] : 0;
        wts[tid]  = v ? wsf[WS_WLIST  + base + gr] : 0.f;
    }
    __syncthreads();

    int lane = tid & 31, wave = tid >> 5;
    int wm = wave & 3, wn = wave >> 2;        // wave tile 32(M) x 64(N)
    int l15 = lane & 15, lh = lane >> 4;

    const float* wb = ew + (size_t)expert * FDIM * DDIM + (size_t)nBase * DDIM;

    v8f accv[2][4];
    v8f vzero = {0.f,0.f,0.f,0.f,0.f,0.f,0.f,0.f};
#pragma unroll
    for (int i = 0; i < 2; ++i)
#pragma unroll
        for (int j = 0; j < 4; ++j) accv[i][j] = vzero;

    // stage one 128x64 K-tile of A and B into LDS buffer `buf`
    auto stage = [&](int buf, int kt) {
        int k0 = kt * BK;
#pragma unroll
        for (int it = 0; it < 8; ++it) {
            int li  = tid + it * 256;         // 0..2047
            int row = li >> 4, seg = li & 15; // 16 float4 per 64-wide row
            const float4* pa = (const float4*)(x + (size_t)toks[row] * DDIM + k0) + seg;
            float4 va = *pa;
            v4bf ba = { (__bf16)va.x, (__bf16)va.y, (__bf16)va.z, (__bf16)va.w };
            *(v4bf*)(&As[buf][row * LDS_STRIDE + seg * 4]) = ba;
            const float4* pb = (const float4*)(wb + (size_t)row * DDIM + k0) + seg;
            float4 vb = *pb;
            v4bf bbv = { (__bf16)vb.x, (__bf16)vb.y, (__bf16)vb.z, (__bf16)vb.w };
            *(v4bf*)(&Bs[buf][row * LDS_STRIDE + seg * 4]) = bbv;
            if (kt + 2 < KTILES) {            // look two tiles ahead
                __builtin_prefetch((const void*)(pa + 32), 0, 2);
                __builtin_prefetch((const void*)(pb + 32), 0, 2);
            }
        }
    };

    stage(0, 0);

    for (int kt = 0; kt < KTILES; ++kt) {
        __syncthreads();
        int buf = kt & 1;
        if (kt + 1 < KTILES) stage(buf ^ 1, kt + 1);

#pragma unroll
        for (int ks = 0; ks < 2; ++ks) {      // two 16x16x32 K-steps per tile
            int koff = ks * 32;
            // A frag: lane<16 -> K 0..7 & 16..23 ; lane>=16 -> K 8..15 & 24..31
            v16bf afr[2], bfr[4];
#pragma unroll
            for (int mi = 0; mi < 2; ++mi) {
                const __bf16* pa = &As[buf][(wm*32 + mi*16 + l15) * LDS_STRIDE + koff + lh*8];
                v8bf lo = *(const v8bf*)pa;
                v8bf hi = *(const v8bf*)(pa + 16);
                afr[mi] = __builtin_shufflevector(lo, hi,
                          0,1,2,3,4,5,6,7,8,9,10,11,12,13,14,15);
            }
            // B frag: lane<16 -> col=l15, K 0..7 & 16..23 ; lane>=16 -> K 8..15 & 24..31
#pragma unroll
            for (int nj = 0; nj < 4; ++nj) {
                const __bf16* pb = &Bs[buf][(wn*64 + nj*16 + l15) * LDS_STRIDE + koff + lh*8];
                v8bf lo = *(const v8bf*)pb;
                v8bf hi = *(const v8bf*)(pb + 16);
                bfr[nj] = __builtin_shufflevector(lo, hi,
                          0,1,2,3,4,5,6,7,8,9,10,11,12,13,14,15);
            }
#pragma unroll
            for (int mi = 0; mi < 2; ++mi)
#pragma unroll
                for (int nj = 0; nj < 4; ++nj)
                    accv[mi][nj] = __builtin_amdgcn_wmma_f32_16x16x32_bf16(
                        false, afr[mi], false, bfr[nj], (short)0, accv[mi][nj],
                        false, false);
        }
    }

    // epilogue: out[tok, n] += w_assign * acc  (hardware f32 atomic; the two
    // expert contributions per token are commutative)
#pragma unroll
    for (int mi = 0; mi < 2; ++mi) {
#pragma unroll
        for (int i = 0; i < 8; ++i) {
            int m = wm*32 + mi*16 + lh*8 + i;
            if (mBase + m < cnt) {
                int   t = toks[m];
                float w = wts[m];
                float* po = out + (size_t)t * FDIM + nBase + wn*64 + l15;
#pragma unroll
                for (int nj = 0; nj < 4; ++nj)
                    unsafeAtomicAdd(po + nj * 16, w * accv[mi][nj][i]);
            }
        }
    }
}

// usage/load-balance loss + utilization
__global__ void __launch_bounds__(256) moe_usage(
    const float* __restrict__ rwout, const int* __restrict__ wsi,
    float* __restrict__ out)
{
    __shared__ float red[8][256];
    int tid = threadIdx.x;
    float acc[8];
#pragma unroll
    for (int e = 0; e < 8; ++e) acc[e] = 0.f;
    for (int t = tid; t < TOKENS; t += 256) {
        const float* r = rwout + (size_t)t * 8;
#pragma unroll
        for (int e = 0; e < 8; ++e) acc[e] += r[e];
    }
#pragma unroll
    for (int e = 0; e < 8; ++e) red[e][tid] = acc[e];
    __syncthreads();
    for (int s = 128; s > 0; s >>= 1) {
        if (tid < s) {
#pragma unroll
            for (int e = 0; e < 8; ++e) red[e][tid] += red[e][tid + s];
        }
        __syncthreads();
    }
    if (tid < 8) out[OUT_UTIL + tid] = (float)wsi[WS_COUNTS + tid] / (float)TOKENS;
    if (tid == 0) {
        float loss = 0.f;
#pragma unroll
        for (int e = 0; e < 8; ++e) {
            float u = red[e][0] / (float)TOKENS - 0.125f;
            loss += u * u;
        }
        out[OUT_LBL] = loss * 0.125f;
    }
}

// per-pair squared L2 over concat(expert_w, expert_b, expert_pref)
__global__ void __launch_bounds__(256) moe_pairdist(
    const float* __restrict__ ew, const float* __restrict__ eb,
    const float* __restrict__ ep, float* __restrict__ psum)
{
    __shared__ float red[256];
    int p = blockIdx.x;
    int i = 0, rem = p;
    while (rem >= 7 - i) { rem -= 7 - i; ++i; }
    int j = i + 1 + rem;
    const int PW = FDIM * DDIM;
    const int P  = PW + 1024 + 1024;
    int tid = threadIdx.x;
    float s = 0.f;
    for (int idx = tid; idx < P; idx += 256) {
        float a, b;
        if (idx < PW)            { a = ew[(size_t)i*PW + idx]; b = ew[(size_t)j*PW + idx]; }
        else if (idx < PW + 1024){ int q = idx - PW;        a = eb[i*1024+q]; b = eb[j*1024+q]; }
        else                     { int q = idx - PW - 1024; a = ep[i*1024+q]; b = ep[j*1024+q]; }
        float d = a - b; s += d * d;
    }
    red[tid] = s; __syncthreads();
    for (int st = 128; st > 0; st >>= 1) {
        if (tid < st) red[tid] += red[tid + st];
        __syncthreads();
    }
    if (tid == 0) psum[p] = red[0];
}

__global__ void moe_divfin(const float* __restrict__ psum, float* __restrict__ out) {
    if (threadIdx.x == 0) {
        float s = 0.f;
        for (int p = 0; p < 28; ++p) s += sqrtf(psum[p]);
        out[OUT_DIV] = s / 28.f;
    }
}

// =======================================================================
extern "C" void kernel_launch(void* const* d_in, const int* in_sizes, int n_in,
                              void* d_out, int out_size, void* d_ws, size_t ws_size,
                              hipStream_t stream) {
    const float* x  = (const float*)d_in[0];
    const float* rw = (const float*)d_in[1];
    const float* rb = (const float*)d_in[2];
    const float* ew = (const float*)d_in[3];
    const float* eb = (const float*)d_in[4];
    const float* ep = (const float*)d_in[5];
    float* out = (float*)d_out;
    int*   wsi = (int*)d_ws;
    float* wsf = (float*)d_ws;

    moe_zero   <<<1, 32, 0, stream>>>(wsi);
    moe_router <<<TOKENS / 8, 256, 0, stream>>>(x, rw, rb, out + OUT_RW, wsi, wsf);
    moe_offsets<<<1, 32, 0, stream>>>(wsi);
    moe_scatter<<<TOKENS / 256, 256, 0, stream>>>(wsi, wsf);
    moe_bias_init<<<TOKENS, 256, 0, stream>>>(eb, wsi, wsf, out);
    dim3 g(TOKENS / 128, FDIM / 128, NEXP);
    moe_gemm   <<<g, 256, 0, stream>>>(x, ew, wsi, wsf, out);
    moe_usage  <<<1, 256, 0, stream>>>(out + OUT_RW, wsi, out);
    moe_pairdist<<<28, 256, 0, stream>>>(ew, eb, ep, wsf + WS_PSUM);
    moe_divfin <<<1, 32, 0, stream>>>(wsf + WS_PSUM, out);
}
